// TGCNAttention_54477365183234
// MI455X (gfx1250) — matrix operand order repaired and checked by reading, not compiled, not run on hardware
//
#include <hip/hip_runtime.h>
#include <hip/hip_bf16.h>

// Problem constants (from reference)
#define BB 16
#define TT 64
#define NN 512
#define HH 128
#define PP 12
#define KC 160          // H+1 = 129 padded to multiple of 32 for bf16 WMMA K-steps
#define O1 (2*HH)       // 256
#define O2 HH           // 128

typedef __attribute__((ext_vector_type(16))) __bf16          v16bf;
typedef __attribute__((ext_vector_type(8)))  float           v8f;
typedef __attribute__((ext_vector_type(8)))  unsigned short  ushort8;

// ---------- bf16 helpers (RNE) ----------
__device__ __forceinline__ unsigned short f32_to_bf16(float f) {
    union { float f; unsigned u; } x; x.f = f;
    unsigned r = x.u + 0x7FFFu + ((x.u >> 16) & 1u);
    return (unsigned short)(r >> 16);
}

// ---------- WMMA fragment builders (ISA 7.12.2 layouts, wave32) ----------
// A-matrix 16x32 bf16 from a row-major source (leading dim ld):
//   lane L: M = L%16, half = L/16
//   elements 0..7  -> K = half*8 + 0..7        (contiguous, 16B)
//   elements 8..15 -> K = 16 + half*8 + 0..7   (contiguous, 16B)
// => two aligned b128 loads per lane, zero shuffling.
__device__ __forceinline__ v16bf load_a_frag(const unsigned short* __restrict__ src,
                                             int ld, int lane) {
    union { v16bf v; ushort8 h[2]; } f;
    const int m = lane & 15, half = lane >> 4;
    const unsigned short* row = src + m * ld + half * 8;
    f.h[0] = *(const ushort8*)(row);
    f.h[1] = *(const ushort8*)(row + 16);
    return f.v;
}

// B-matrix 32x16 bf16 from a TRANSPOSED source (rows = output col, cols = K,
// leading dim ld):  lane L: N = L%16, half = L/16 ; elements 0..7: K = half*8 + e,
// elements 8..15: K = 16 + half*8 + (e-8)  => two aligned b128 loads.
__device__ __forceinline__ v16bf load_b_frag_tr(const unsigned short* __restrict__ src,
                                                int ld, int lane) {
    union { v16bf v; ushort8 h[2]; } f;
    const int n = lane & 15, half = lane >> 4;
    const unsigned short* p = src + n * ld + half * 8;
    f.h[0] = *(const ushort8*)(p);
    f.h[1] = *(const ushort8*)(p + 16);
    return f.v;
}

__device__ __forceinline__ v8f wmma_bf16(v16bf a, v16bf b, v8f c) {
    return __builtin_amdgcn_wmma_f32_16x16x32_bf16(
               false, a, false, b, (short)0, c, false, false);
}

// ---------- prep: cast f32 [K,O] -> bf16 [Kp,O], zero-padding rows K..Kp -----
__global__ void castpad_kernel(const float* __restrict__ src,
                               unsigned short* __restrict__ dst,
                               int K, int Kp, int O) {
    int i = blockIdx.x * blockDim.x + threadIdx.x;
    if (i >= Kp * O) return;
    int k = i / O, o = i % O;
    dst[i] = f32_to_bf16(k < K ? src[k * O + o] : 0.0f);
}

// ---------- prep: cast+TRANSPOSE f32 [K,O] -> bf16 [O,Kp] (zero pad K..Kp) ---
__global__ void castpad_t_kernel(const float* __restrict__ src,
                                 unsigned short* __restrict__ dst,
                                 int K, int Kp, int O) {
    int i = blockIdx.x * blockDim.x + threadIdx.x;
    if (i >= O * Kp) return;
    int o = i / Kp, k = i % Kp;
    dst[i] = f32_to_bf16(k < K ? src[k * O + o] : 0.0f);
}

// ---------- build conc = [x_t | h] (or [x_t | r*h]) as bf16 [B,N,KC] ----------
__global__ void concat1_kernel(const float* __restrict__ x,   // [B,T,N]
                               const float* __restrict__ h,   // [B,N*H]
                               unsigned short* __restrict__ conc, int t) {
    int i = blockIdx.x * blockDim.x + threadIdx.x;
    if (i >= BB * NN * KC) return;
    int c = i % KC, bn = i / KC, n = bn % NN, b = bn / NN;
    float v = 0.0f;
    if (c == 0)        v = x[(b * TT + t) * NN + n];
    else if (c <= HH)  v = h[b * (NN * HH) + n * HH + (c - 1)];
    conc[i] = f32_to_bf16(v);
}

__global__ void concat2_kernel(const float* __restrict__ x,
                               const float* __restrict__ h,
                               const float* __restrict__ ru,  // sigmoid'd [B,N*2H]
                               unsigned short* __restrict__ conc, int t) {
    int i = blockIdx.x * blockDim.x + threadIdx.x;
    if (i >= BB * NN * KC) return;
    int c = i % KC, bn = i / KC, n = bn % NN, b = bn / NN;
    float v = 0.0f;
    if (c == 0)       v = x[(b * TT + t) * NN + n];
    else if (c <= HH) {
        int idx = n * HH + (c - 1);                 // flat index into [N*H]
        v = ru[b * (NN * 2 * HH) + idx]             // r = first half of flat ru
          * h[b * (NN * HH) + idx];
    }
    conc[i] = f32_to_bf16(v);
}

// ---------- GEMM1: agg[b] = L @ conc[b] ; M=512,K=512,cols=KC ; bf16 out ------
// Block = 128 threads = 4 waves covering a 64x32 output tile; each wave owns a
// 16-row strip and BOTH 16-col halves (2 accumulators sharing one A fragment):
// per K-step a wave issues 2 A-loads + 4 B-loads (b128) for 2 WMMAs.
__global__ __launch_bounds__(128) void gemm_lap_kernel(
        const unsigned short* __restrict__ Lbf,   // [512,512] bf16
        const unsigned short* __restrict__ X,     // [B,512,KC] bf16
        unsigned short* __restrict__ Y) {         // [B,512,KC] bf16
    __shared__ __align__(16) unsigned short ldsA[64 * 32];   // [m][k]
    __shared__ __align__(16) unsigned short ldsBT[32 * 32];  // [col][k]
    const int wave = threadIdx.x >> 5, lane = threadIdx.x & 31;
    const int c0 = blockIdx.x * 32;        // 0..4  -> col tile base
    const int m0 = blockIdx.y * 64;        // 0..7  -> row tile base
    const int b  = blockIdx.z;
    const int wm = wave * 16;              // wave's row strip inside 64
    const unsigned short* Xb = X + b * (NN * KC);
    v8f acc0 = {}, acc1 = {};
    for (int k0 = 0; k0 < NN; k0 += 32) {
        // prefetch next K tile while staging this one (global_prefetch_b8)
        if (k0 + 32 < NN) {
            __builtin_prefetch(Lbf + (m0 + (threadIdx.x & 63)) * NN + k0 + 32, 0, 1);
            __builtin_prefetch(Xb + (k0 + 32 + (threadIdx.x & 31)) * KC + c0, 0, 1);
        }
        // stage A 64x32 row-major (1024 dwords, 8 per thread)
        for (int i = threadIdx.x; i < 1024; i += 128) {
            int r = i >> 4, q = i & 15;
            ((unsigned*)ldsA)[i] =
                *(const unsigned*)(Lbf + (m0 + r) * NN + k0 + q * 2);
        }
        // stage B 32x32 transposed (512 dwords read, scattered as b16 pairs)
        for (int i = threadIdx.x; i < 512; i += 128) {
            int r = i >> 4, q = i & 15;
            unsigned w =
                *(const unsigned*)(Xb + (k0 + r) * KC + c0 + q * 2);
            ldsBT[(q * 2)     * 32 + r] = (unsigned short)(w & 0xFFFFu);
            ldsBT[(q * 2 + 1) * 32 + r] = (unsigned short)(w >> 16);
        }
        __syncthreads();
        v16bf af  = load_a_frag(ldsA + wm * 32, 32, lane);
        v16bf bf0 = load_b_frag_tr(ldsBT, 32, lane);
        v16bf bf1 = load_b_frag_tr(ldsBT + 16 * 32, 32, lane);
        acc0 = wmma_bf16(af, bf0, acc0);
        acc1 = wmma_bf16(af, bf1, acc1);
        __syncthreads();
    }
    unsigned short* Yb = Y + b * (NN * KC);
    const int row = (lane >> 4) * 8, col = lane & 15;
#pragma unroll
    for (int g = 0; g < 8; ++g) {
        Yb[(m0 + wm + row + g) * KC + (c0 + col)]      = f32_to_bf16(acc0[g]);
        Yb[(m0 + wm + row + g) * KC + (c0 + 16 + col)] = f32_to_bf16(acc1[g]);
    }
}

// ---------- GEMM2: out[b] = act(agg[b] @ W + bias) ; K=KC, fused epilogue -----
// A read row-major (two b128 per lane, shared by 2 accumulators); W
// pre-transposed [O,KC]. Each wave computes a 16x32 output tile.
__global__ __launch_bounds__(128) void gemm_w_kernel(
        const unsigned short* __restrict__ A,   // [B,512,KC] bf16
        const unsigned short* __restrict__ Wt,  // [O,KC] bf16 (transposed, padded)
        const float* __restrict__ bias,         // [O] f32
        float* __restrict__ out,                // [B,512,O] f32
        int O, int act) {                       // act: 0=sigmoid, 1=tanh
    const int wave = threadIdx.x >> 5, lane = threadIdx.x & 31;
    const int c0 = blockIdx.x * 128 + wave * 32;
    const int m0 = blockIdx.y * 16;
    const int b  = blockIdx.z;
    const unsigned short* Ab = A + b * (NN * KC) + m0 * KC;
    v8f acc0 = {}, acc1 = {};
#pragma unroll
    for (int k0 = 0; k0 < KC; k0 += 32) {
        v16bf af  = load_a_frag(Ab + k0, KC, lane);
        v16bf bf0 = load_b_frag_tr(Wt + c0 * KC + k0, KC, lane);
        v16bf bf1 = load_b_frag_tr(Wt + (c0 + 16) * KC + k0, KC, lane);
        acc0 = wmma_bf16(af, bf0, acc0);
        acc1 = wmma_bf16(af, bf1, acc1);
    }
    float* ob = out + b * (NN * O);
    const int row = (lane >> 4) * 8, col = lane & 15;
    const float bv0 = bias[c0 + col];
    const float bv1 = bias[c0 + 16 + col];
#pragma unroll
    for (int g = 0; g < 8; ++g) {
        float v0 = acc0[g] + bv0;
        float v1 = acc1[g] + bv1;
        v0 = (act == 0) ? (1.0f / (1.0f + expf(-v0))) : tanhf(v0);
        v1 = (act == 0) ? (1.0f / (1.0f + expf(-v1))) : tanhf(v1);
        ob[(m0 + row + g) * O + (c0 + col)]      = v0;
        ob[(m0 + row + g) * O + (c0 + 16 + col)] = v1;
    }
}

// ---------- GRU update fused with hs1 = h_new . att_w1 + att_b1 --------------
__global__ __launch_bounds__(128) void gru_kernel(
        const float* __restrict__ ru,    // [B,N*2H] (sigmoid)
        const float* __restrict__ cbuf,  // [B,N*H]  (tanh)
        float* __restrict__ h,           // [B,N*H]  in/out
        const float* __restrict__ att_w1, const float* __restrict__ att_b1,
        float* __restrict__ hs1, int t) {
    __shared__ float red[128];
    const int bn = blockIdx.x, n = bn % NN, b = bn / NN, c = threadIdx.x;
    const int idx = n * HH + c;
    float u  = ru[b * (NN * 2 * HH) + NN * HH + idx];   // u = second flat half
    float hv = h[b * (NN * HH) + idx];
    float cv = cbuf[b * (NN * HH) + idx];
    float hn = u * hv + (1.0f - u) * cv;
    h[b * (NN * HH) + idx] = hn;
    red[c] = hn * att_w1[c];
    __syncthreads();
#pragma unroll
    for (int s = 64; s > 0; s >>= 1) {
        if (c < s) red[c] += red[c + s];
        __syncthreads();
    }
    if (c == 0) hs1[(b * TT + t) * NN + n] = red[0] + att_b1[0];
}

// ---------- attention: beta = softmax_T((hs1@w3+b3)*(hs1@w2+b2)) --------------
__global__ __launch_bounds__(64) void beta_kernel(
        const float* __restrict__ hs1,
        const float* __restrict__ w2, const float* __restrict__ w3,
        const float* __restrict__ b2, const float* __restrict__ b3,
        float* __restrict__ beta) {
    __shared__ float sv[TT];
    const int b = blockIdx.x, t = threadIdx.x;
    const float* hrow = hs1 + (b * TT + t) * NN;
    float f = 0.0f, g = 0.0f;
    for (int n = 0; n < NN; ++n) { float v = hrow[n]; f += v * w2[n]; g += v * w3[n]; }
    f += b2[0]; g += b3[0];
    float s = f * g;
    sv[t] = s; __syncthreads();
    float m = sv[0];
    for (int i = 1; i < TT; ++i) m = fmaxf(m, sv[i]);
    __syncthreads();
    float e = expf(s - m);
    sv[t] = e; __syncthreads();
    float sum = 0.0f;
    for (int i = 0; i < TT; ++i) sum += sv[i];
    beta[b * TT + t] = e / sum;
}

// ---------- tail: out[b,p,n] = sum_t beta*hs1*W_out + b_out ------------------
__global__ void output_kernel(const float* __restrict__ hs1,
                              const float* __restrict__ beta,
                              const float* __restrict__ W_out,
                              const float* __restrict__ b_out,
                              float* __restrict__ out) {
    int i = blockIdx.x * blockDim.x + threadIdx.x;     // over B*P*N
    if (i >= BB * PP * NN) return;
    int n = i % NN, bp = i / NN, p = bp % PP, b = bp / PP;
    float acc = b_out[p];
    for (int t = 0; t < TT; ++t)
        acc += beta[b * TT + t] * hs1[(b * TT + t) * NN + n] * W_out[t * PP + p];
    out[i] = acc;
}

// ============================ host launcher ==================================
extern "C" void kernel_launch(void* const* d_in, const int* in_sizes, int n_in,
                              void* d_out, int out_size, void* d_ws, size_t ws_size,
                              hipStream_t stream) {
    const float* x      = (const float*)d_in[0];   // [B,T,N]
    const float* lap    = (const float*)d_in[1];   // [N,N]
    const float* W1     = (const float*)d_in[2];   // [H+1,2H]
    const float* b1     = (const float*)d_in[3];
    const float* W2     = (const float*)d_in[4];   // [H+1,H]
    const float* b2     = (const float*)d_in[5];
    const float* att_w1 = (const float*)d_in[6];   // [H,1]
    const float* att_w2 = (const float*)d_in[7];   // [N,1]
    const float* att_w3 = (const float*)d_in[8];
    const float* att_b1 = (const float*)d_in[9];
    const float* att_b2 = (const float*)d_in[10];
    const float* att_b3 = (const float*)d_in[11];
    const float* W_out  = (const float*)d_in[12];  // [T,P]
    const float* b_out  = (const float*)d_in[13];

    // carve workspace (~24 MB)
    char* p = (char*)d_ws;
    auto carve = [&](size_t bytes) { void* r = p; p += (bytes + 255) & ~(size_t)255; return r; };
    float*          h     = (float*)         carve(sizeof(float) * BB * NN * HH);
    unsigned short* Lbf   = (unsigned short*)carve(sizeof(unsigned short) * NN * NN);
    unsigned short* W1t   = (unsigned short*)carve(sizeof(unsigned short) * O1 * KC);
    unsigned short* W2t   = (unsigned short*)carve(sizeof(unsigned short) * O2 * KC);
    unsigned short* conc  = (unsigned short*)carve(sizeof(unsigned short) * BB * NN * KC);
    unsigned short* agg   = (unsigned short*)carve(sizeof(unsigned short) * BB * NN * KC);
    float*          ru    = (float*)         carve(sizeof(float) * BB * NN * 2 * HH);
    float*          cbuf  = (float*)         carve(sizeof(float) * BB * NN * HH);
    float*          hs1   = (float*)         carve(sizeof(float) * BB * TT * NN);
    float*          beta  = (float*)         carve(sizeof(float) * BB * TT);

    // prep: bf16 Laplacian, transposed zero-padded bf16 weights, zero hidden
    castpad_kernel<<<(NN * NN + 255) / 256, 256, 0, stream>>>(lap, Lbf, NN, NN, NN);
    castpad_t_kernel<<<(O1 * KC + 255) / 256, 256, 0, stream>>>(W1, W1t, HH + 1, KC, O1);
    castpad_t_kernel<<<(O2 * KC + 255) / 256, 256, 0, stream>>>(W2, W2t, HH + 1, KC, O2);
    hipMemsetAsync(h, 0, sizeof(float) * BB * NN * HH, stream);

    const int nConc = (BB * NN * KC + 255) / 256;
    for (int t = 0; t < TT; ++t) {
        // gate path: ru = sigmoid(L@[x|h] @ W1 + b1)
        concat1_kernel<<<nConc, 256, 0, stream>>>(x, h, conc, t);
        gemm_lap_kernel<<<dim3(KC / 32, NN / 64, BB), 128, 0, stream>>>(Lbf, conc, agg);
        gemm_w_kernel<<<dim3(O1 / 128, NN / 16, BB), 128, 0, stream>>>(agg, W1t, b1, ru, O1, 0);
        // candidate path: c = tanh(L@[x|r*h] @ W2 + b2)
        concat2_kernel<<<nConc, 256, 0, stream>>>(x, h, ru, conc, t);
        gemm_lap_kernel<<<dim3(KC / 32, NN / 64, BB), 128, 0, stream>>>(Lbf, conc, agg);
        gemm_w_kernel<<<dim3(O2 / 128, NN / 16, BB), 128, 0, stream>>>(agg, W2t, b2, cbuf, O2, 1);
        // GRU update + fused att_w1 projection
        gru_kernel<<<BB * NN, 128, 0, stream>>>(ru, cbuf, h, att_w1, att_b1, hs1, t);
    }

    beta_kernel<<<BB, 64, 0, stream>>>(hs1, att_w2, att_w3, att_b2, att_b3, beta);
    output_kernel<<<(BB * PP * NN + 255) / 256, 256, 0, stream>>>(
        hs1, beta, W_out, b_out, (float*)d_out);
}
